// DalleSelfAttention_40948218200174
// MI455X (gfx1250) — matrix operand order repaired and checked by reading, not compiled, not run on hardware
//
#include <hip/hip_runtime.h>
#include <hip/hip_bf16.h>

typedef __attribute__((ext_vector_type(16))) __bf16 v16bf;
typedef __attribute__((ext_vector_type(8)))  __bf16 v8bf;
typedef __attribute__((ext_vector_type(8)))  float  v8f;
typedef __attribute__((ext_vector_type(4))) unsigned int u32x4;
typedef __attribute__((ext_vector_type(8))) int         i32x8;
typedef __attribute__((ext_vector_type(4))) int         i32x4;

constexpr int   kBatch = 2;
constexpr int   kSeq   = 2048;
constexpr int   kHid   = 2048;
constexpr int   kHeads = 16;
constexpr int   kHd    = 128;                    // head dim
constexpr int   kK     = 2048;                   // GEMM K dim (= kHid)
// score scale folded with log2(e): softmax computed in base-2 domain
constexpr float kScaleL2 = 0.08838834764831845f * 1.4426950408889634f;
constexpr float kMaskL2  = -14427.0f;            // -10000 * log2(e)

static __device__ inline v16bf make16(v8bf lo, v8bf hi) {
  union { v16bf v; v8bf h[2]; } u;
  u.h[0] = lo; u.h[1] = hi;
  return u.v;
}
static __device__ inline v8f v8f_zero() {
  v8f z;
#pragma unroll
  for (int i = 0; i < 8; ++i) z[i] = 0.0f;
  return z;
}
static __device__ inline v8f wmma_bf16(v16bf a, v16bf b, v8f c) {
  return __builtin_amdgcn_wmma_f32_16x16x32_bf16(false, a, false, b,
                                                 (short)0, c, false, false);
}
static __device__ inline unsigned lds_off(const void* p) {
  return (unsigned)(unsigned long long)p;   // low 32 bits = LDS byte offset
}
static __device__ inline float fast_exp2(float x) {
#if __has_builtin(__builtin_amdgcn_exp2f)
  return __builtin_amdgcn_exp2f(x);         // v_exp_f32 (native base-2)
#else
  return exp2f(x);
#endif
}

// ---------------------------------------------------------------------------
// TDM: async 2D tile load global->LDS (bf16 elements), row padding in LDS.
// tensor_d0 = row length (contig, elems), tensor_d1 = #rows, stride0 in elems.
// tile_d0 x tile_d1 tile at gaddr. pad: every (8B << padi) bytes stored,
// insert (pada+1) DWORDs of padding into the LDS destination address.
// ---------------------------------------------------------------------------
static __device__ inline void tdm_load_2d_bf16(
    unsigned lds_byte_off, const void* gaddr,
    unsigned tensor_d0, unsigned tensor_d1, unsigned long long stride0,
    unsigned tile_d0, unsigned tile_d1,
    unsigned padi, unsigned pada)
{
  const unsigned long long ga = (unsigned long long)gaddr;
  u32x4 g0;
  g0[0] = 1u;                                            // count=1 (user D#)
  g0[1] = lds_byte_off;                                  // lds_addr
  g0[2] = (unsigned)(ga & 0xffffffffu);                  // global_addr lo
  g0[3] = (unsigned)((ga >> 32) & 0x1ffffffu) | (2u << 30);  // hi + type=2
  i32x8 g1;
  g1[0] = (int)((1u << 16)                               // data_size = 2B
              | (1u << 20)                               // pad_enable
              | (padi << 22) | (pada << 25));
  g1[1] = (int)((tensor_d0 & 0xffffu) << 16);
  g1[2] = (int)(((tensor_d0 >> 16) & 0xffffu) | ((tensor_d1 & 0xffffu) << 16));
  g1[3] = (int)(((tensor_d1 >> 16) & 0xffffu) | ((tile_d0 & 0xffffu) << 16));
  g1[4] = (int)(tile_d1 & 0xffffu);                      // tile_dim1 (dim2=0)
  g1[5] = (int)(unsigned)(stride0 & 0xffffffffu);
  g1[6] = (int)(unsigned)((stride0 >> 32) & 0xffffu);    // stride0 hi, dim1_stride=0
  g1[7] = 0;
  const i32x4 z4 = {0, 0, 0, 0};
  const i32x8 z8 = {0, 0, 0, 0, 0, 0, 0, 0};
  __builtin_amdgcn_tensor_load_to_lds(g0, g1, z4, z4, z8, 0);
}

// ---------------------------------------------------------------------------
// Online-softmax update for one 16x32 score tile held as two C-layout v8f.
// MASK=false for interior tiles (causal always valid) skips compare/cndmask.
// Base-2 domain throughout. Also writes P (bf16) to LDS row-major.
// ---------------------------------------------------------------------------
template <bool MASK>
static __device__ inline void softmax_update(
    const v8f& s0, const v8f& s1, int m0, int t0, int l16, int hi,
    float* rmax, float* rsum, v8f* o, __bf16* pl)
{
#pragma unroll
  for (int r = 0; r < 8; ++r) {
    float v0 = s0[r] * kScaleL2;
    float v1 = s1[r] * kScaleL2;
    if (MASK) {
      const int mrow = m0 + r + 8 * hi;
      if (t0 + l16 > mrow)      v0 = kMaskL2;
      if (t0 + 16 + l16 > mrow) v1 = kMaskL2;
    }
    float tm = fmaxf(v0, v1);
#pragma unroll
    for (int off = 1; off < 16; off <<= 1)
      tm = fmaxf(tm, __shfl_xor(tm, off, 32));
    const float nm = fmaxf(rmax[r], tm);
    const float sf = fast_exp2(rmax[r] - nm);
    rmax[r] = nm;
    const float p0e = fast_exp2(v0 - nm);
    const float p1e = fast_exp2(v1 - nm);
    float ps = p0e + p1e;
#pragma unroll
    for (int off = 1; off < 16; off <<= 1)
      ps += __shfl_xor(ps, off, 32);
    rsum[r] = rsum[r] * sf + ps;
#pragma unroll
    for (int d = 0; d < 8; ++d) o[d][r] *= sf;
    const int prow = r + 8 * hi;
    pl[prow * 40 + l16]      = (__bf16)p0e;   // C-layout -> LDS row-major
    pl[prow * 40 + 16 + l16] = (__bf16)p1e;
  }
}

// ---------------------------------------------------------------------------
// Tiled bf16 WMMA GEMM: C[M=4096][N] = A[M][2048] * Bw[N][2048]^T + bias
// Block tile 128(M) x 256(N), 8 waves, each wave 64x64 (4x4 WMMA).
// LDS row stride 40 elems (20 DW, gcd(20,64)=4 -> conflict-optimal b128 reads)
// A_BF16: A tile DMA'd via TDM. EPI==0: fp32 out. EPI==1: route bf16 Q/K/V^T.
// ---------------------------------------------------------------------------
template <bool A_BF16, int EPI>
__global__ __launch_bounds__(256)
void gemm_bf16_wmma(const void*  __restrict__ Aab,
                    const float* __restrict__ Bw,
                    const float* __restrict__ bias,
                    float*  __restrict__ OutF,
                    __bf16* __restrict__ Qb,
                    __bf16* __restrict__ Kb,
                    __bf16* __restrict__ Vb)
{
  __shared__ __attribute__((aligned(64))) __bf16 As[128 * 40];
  __shared__ __attribute__((aligned(64))) __bf16 Bs[256 * 40];

  const int tid  = threadIdx.x;
  const int lane = tid & 31;
  const int wid  = tid >> 5;
  const int l16  = lane & 15;
  const int hi   = lane >> 4;
  const int wm   = wid & 1;    // 64-row slab
  const int wn   = wid >> 1;   // 64-col slab
  const int blkN = blockIdx.x;
  const int blkM = blockIdx.y;

  v8f acc[4][4];
#pragma unroll
  for (int mt = 0; mt < 4; ++mt)
#pragma unroll
    for (int nt = 0; nt < 4; ++nt) acc[mt][nt] = v8f_zero();

  const int  arow = tid >> 1;         // 0..127
  const int  acol = (tid & 1) * 16;   // 0 / 16
  const long gmA  = (long)blkM * 128 + arow;
  const long gnB  = (long)blkN * 256 + tid;   // B row = n

  for (int k0 = 0; k0 < kK; k0 += 32) {
    __syncthreads();   // previous iteration's fragments consumed
    if constexpr (A_BF16) {
      // TDM: 32x128 tile of bf16 A, rows land at stride 40 elems (pad 4 DW/16 DW)
      if (wid == 0) {
        const __bf16* Ab = (const __bf16*)Aab;
        tdm_load_2d_bf16(lds_off(&As[0]),
                         (const void*)(Ab + (size_t)blkM * 128 * kK + k0),
                         kK, 4096, kK, 32, 128, /*padi=*/3, /*pada=*/3);
      }
    } else {
      const float*  Af  = (const float*)Aab;
      const float4* src = (const float4*)(Af + gmA * kK + k0 + acol);
      float4 f0 = src[0], f1 = src[1], f2 = src[2], f3 = src[3];
      __builtin_prefetch((const void*)(src + 8), 0, 1);
      v8bf p0 = { (__bf16)f0.x,(__bf16)f0.y,(__bf16)f0.z,(__bf16)f0.w,
                  (__bf16)f1.x,(__bf16)f1.y,(__bf16)f1.z,(__bf16)f1.w };
      v8bf p1 = { (__bf16)f2.x,(__bf16)f2.y,(__bf16)f2.z,(__bf16)f2.w,
                  (__bf16)f3.x,(__bf16)f3.y,(__bf16)f3.z,(__bf16)f3.w };
      __bf16* dst = &As[arow * 40 + acol];
      *(v8bf*)dst       = p0;
      *(v8bf*)(dst + 8) = p1;
    }
    { // stage B tile 256x32 from fp32 weights (row n, K-contiguous)
      const float4* src = (const float4*)(Bw + gnB * kK + k0);
      float4 f0 = src[0], f1 = src[1], f2 = src[2], f3 = src[3];
      float4 f4 = src[4], f5 = src[5], f6 = src[6], f7 = src[7];
      __builtin_prefetch((const void*)(src + 8), 0, 1);
      __bf16* dst = &Bs[tid * 40];
      v8bf p0 = { (__bf16)f0.x,(__bf16)f0.y,(__bf16)f0.z,(__bf16)f0.w,
                  (__bf16)f1.x,(__bf16)f1.y,(__bf16)f1.z,(__bf16)f1.w };
      v8bf p1 = { (__bf16)f2.x,(__bf16)f2.y,(__bf16)f2.z,(__bf16)f2.w,
                  (__bf16)f3.x,(__bf16)f3.y,(__bf16)f3.z,(__bf16)f3.w };
      v8bf p2 = { (__bf16)f4.x,(__bf16)f4.y,(__bf16)f4.z,(__bf16)f4.w,
                  (__bf16)f5.x,(__bf16)f5.y,(__bf16)f5.z,(__bf16)f5.w };
      v8bf p3 = { (__bf16)f6.x,(__bf16)f6.y,(__bf16)f6.z,(__bf16)f6.w,
                  (__bf16)f7.x,(__bf16)f7.y,(__bf16)f7.z,(__bf16)f7.w };
      *(v8bf*)(dst)      = p0;
      *(v8bf*)(dst + 8)  = p1;
      *(v8bf*)(dst + 16) = p2;
      *(v8bf*)(dst + 24) = p3;
    }
    if constexpr (A_BF16) {
      if (wid == 0) __builtin_amdgcn_s_wait_tensorcnt(0);
    }
    __syncthreads();

    v16bf af[4], bfr[4];
#pragma unroll
    for (int mt = 0; mt < 4; ++mt) {
      const __bf16* p = &As[(wm * 64 + mt * 16 + l16) * 40 + hi * 8];
      af[mt] = make16(*(const v8bf*)p, *(const v8bf*)(p + 16));
    }
#pragma unroll
    for (int nt = 0; nt < 4; ++nt) {
      const __bf16* p = &Bs[(wn * 64 + nt * 16 + l16) * 40 + hi * 8];
      bfr[nt] = make16(*(const v8bf*)p, *(const v8bf*)(p + 16));
    }
#pragma unroll
    for (int mt = 0; mt < 4; ++mt)
#pragma unroll
      for (int nt = 0; nt < 4; ++nt)
        acc[mt][nt] = wmma_bf16(af[mt], bfr[nt], acc[mt][nt]);
  }

  // epilogue (C layout: vgpr r -> row r + 8*hi, col = l16)
#pragma unroll
  for (int nt = 0; nt < 4; ++nt) {
    const int   n  = blkN * 256 + wn * 64 + nt * 16 + l16;
    const float bn = bias[n];
#pragma unroll
    for (int mt = 0; mt < 4; ++mt) {
#pragma unroll
      for (int r = 0; r < 8; ++r) {
        const int   m   = blkM * 128 + wm * 64 + mt * 16 + r + 8 * hi;
        const float val = acc[mt][nt][r] + bn;
        if constexpr (EPI == 0) {
          OutF[(size_t)m * kHid + n] = val;
        } else {
          const int bidx = m >> 11;        // batch
          const int s    = m & 2047;       // seq pos
          const int sec  = n >> 11;        // 0:Q 1:K 2:V
          const int np   = n & 2047;
          const int hh   = np >> 7;        // head
          const int d    = np & 127;       // head-dim
          const __bf16 bv   = (__bf16)val;
          const size_t head = (size_t)(bidx * kHeads + hh);
          if (sec == 0)      Qb[(head * kSeq + s) * kHd + d] = bv;  // [b,h,s,d]
          else if (sec == 1) Kb[(head * kSeq + s) * kHd + d] = bv;  // [b,h,s,d]
          else               Vb[(head * kHd + d) * kSeq + s] = bv;  // V^T [b,h,d,t]
        }
      }
    }
  }
}

// ---------------------------------------------------------------------------
// Causal flash attention, bf16 WMMA. 8 waves x 16 query rows per block.
// K/V tiles double-buffered in LDS via TDM (one issuing wave, TENSORcnt sync).
// Fragments preloaded in bulk so WMMAs issue back-to-back behind one wait.
// grid = (S/128, B*N), block = 256. ctx written bf16 as [b,s,h].
// ---------------------------------------------------------------------------
__global__ __launch_bounds__(256)
void flash_attn_wmma(const __bf16* __restrict__ Qb,
                     const __bf16* __restrict__ Kb,
                     const __bf16* __restrict__ Vt,
                     __bf16* __restrict__ Ctx)
{
  // K tile: 32 rows x 128 elems, LDS row stride 136 elems (68 DW, rot 4)
  // V^T tile: 128 rows x 32 elems, LDS row stride 40 elems (20 DW, rot 20)
  __shared__ __attribute__((aligned(64))) __bf16 Klds[2][32 * 136];
  __shared__ __attribute__((aligned(64))) __bf16 Vlds[2][128 * 40];
  __shared__ __attribute__((aligned(64))) __bf16 Plds[8][16 * 40];

  const int tid  = threadIdx.x;
  const int lane = tid & 31;
  const int wid  = tid >> 5;
  const int l16  = lane & 15;
  const int hi   = lane >> 4;

  const int bh = blockIdx.y;            // 0..31
  const int b  = bh >> 4;
  const int h  = bh & 15;
  const int m0 = blockIdx.x * 128 + wid * 16;

  const __bf16* qp = Qb + (size_t)bh * kSeq * kHd;
  const __bf16* kp = Kb + (size_t)bh * kSeq * kHd;
  const __bf16* vp = Vt + (size_t)bh * kHd * kSeq;

  // Q fragments (A operand, rows m0..m0+15, K = 128 = 4 chunks) stay in regs
  v16bf qf[4];
#pragma unroll
  for (int c = 0; c < 4; ++c) {
    const __bf16* p = qp + (size_t)(m0 + l16) * kHd + c * 32 + hi * 8;
    qf[c] = make16(*(const v8bf*)p, *(const v8bf*)(p + 16));
  }

  v8f o[8];
#pragma unroll
  for (int d = 0; d < 8; ++d) o[d] = v8f_zero();
  float rmax[8], rsum[8];
#pragma unroll
  for (int r = 0; r < 8; ++r) { rmax[r] = -1e30f; rsum[r] = 0.0f; }

  __bf16* pl = &Plds[wid][0];
  const int ntiles = blockIdx.x * 4 + 4;   // keys 0 .. blk*128+127, 32 per tile

  // prologue: DMA tile 0 (K and V^T) into buffer 0
  if (wid == 0) {
    tdm_load_2d_bf16(lds_off(&Klds[0][0]), (const void*)kp,
                     kHd, kSeq, kHd, 128, 32, /*padi=*/5, /*pada=*/3);
    tdm_load_2d_bf16(lds_off(&Vlds[0][0]), (const void*)vp,
                     kSeq, kHd, kSeq, 32, 128, /*padi=*/3, /*pada=*/3);
    __builtin_amdgcn_s_wait_tensorcnt(0);
  }
  __syncthreads();

  for (int kt = 0; kt < ntiles; ++kt) {
    const int t0  = kt * 32;
    const int cur = kt & 1;

    // issue DMA for next tile into the other buffer (overlaps compute)
    if (wid == 0 && kt + 1 < ntiles) {
      const int tn = (kt + 1) * 32;
      tdm_load_2d_bf16(lds_off(&Klds[1 - cur][0]),
                       (const void*)(kp + (size_t)tn * kHd),
                       kHd, kSeq, kHd, 128, 32, 5, 3);
      tdm_load_2d_bf16(lds_off(&Vlds[1 - cur][0]),
                       (const void*)(vp + tn),
                       kSeq, kHd, kSeq, 32, 128, 3, 3);
    }

    if (t0 <= m0 + 15) {   // causal: this wave still has live keys here
      const __bf16* kb = &Klds[cur][0];
      const __bf16* vb = &Vlds[cur][0];

      // ---- QK: preload all 8 K fragments, then 8 back-to-back WMMAs ----
      v16bf kf[8];
#pragma unroll
      for (int c = 0; c < 4; ++c) {
        const __bf16* p0 = &kb[l16 * 136 + c * 32 + hi * 8];
        const __bf16* p1 = &kb[(16 + l16) * 136 + c * 32 + hi * 8];
        kf[c]     = make16(*(const v8bf*)p0, *(const v8bf*)(p0 + 16));
        kf[4 + c] = make16(*(const v8bf*)p1, *(const v8bf*)(p1 + 16));
      }
      v8f s0 = v8f_zero(), s1 = v8f_zero();
#pragma unroll
      for (int c = 0; c < 4; ++c) s0 = wmma_bf16(qf[c], kf[c], s0);
#pragma unroll
      for (int c = 0; c < 4; ++c) s1 = wmma_bf16(qf[c], kf[4 + c], s1);

      // ---- online softmax (mask only on diagonal tiles) ----
      if (t0 + 31 <= m0)
        softmax_update<false>(s0, s1, m0, t0, l16, hi, rmax, rsum, o, pl);
      else
        softmax_update<true >(s0, s1, m0, t0, l16, hi, rmax, rsum, o, pl);
      asm volatile("s_wait_dscnt 0" ::: "memory");

      // reload P as A operand (16x32 bf16)
      const __bf16* pp = pl + l16 * 40 + hi * 8;
      v16bf pf = make16(*(const v8bf*)pp, *(const v8bf*)(pp + 16));

      // ---- PV: preload all 8 V fragments, then 8 back-to-back WMMAs ----
      v16bf vf[8];
#pragma unroll
      for (int d = 0; d < 8; ++d) {
        const __bf16* vq = &vb[(d * 16 + l16) * 40 + hi * 8];
        vf[d] = make16(*(const v8bf*)vq, *(const v8bf*)(vq + 16));
      }
#pragma unroll
      for (int d = 0; d < 8; ++d) o[d] = wmma_bf16(pf, vf[d], o[d]);
    }

    __syncthreads();                                  // tile `cur` consumed
    if (wid == 0) __builtin_amdgcn_s_wait_tensorcnt(0);  // next tile landed
    __syncthreads();
  }

  // normalize + store ctx bf16 as [b, s, h*128+d]
  float inv[8];
#pragma unroll
  for (int r = 0; r < 8; ++r) inv[r] = 1.0f / rsum[r];
#pragma unroll
  for (int d = 0; d < 8; ++d) {
#pragma unroll
    for (int r = 0; r < 8; ++r) {
      const int mrow = m0 + r + 8 * hi;
      Ctx[((size_t)b * kSeq + mrow) * kHid + h * kHd + d * 16 + l16] =
          (__bf16)(o[d][r] * inv[r]);
    }
  }
}

// ---------------------------------------------------------------------------
extern "C" void kernel_launch(void* const* d_in, const int* in_sizes, int n_in,
                              void* d_out, int out_size, void* d_ws, size_t ws_size,
                              hipStream_t stream)
{
  (void)in_sizes; (void)n_in; (void)out_size; (void)ws_size;
  const float* hidden  = (const float*)d_in[0];
  // d_in[1] = ltor_mask (causal, implicit in the kernel)
  const float* w_qkv   = (const float*)d_in[2];
  const float* b_qkv   = (const float*)d_in[3];
  const float* w_dense = (const float*)d_in[4];
  const float* b_dense = (const float*)d_in[5];
  float* out = (float*)d_out;

  const size_t headElems = (size_t)kBatch * kHeads * kSeq * kHd;  // 8M
  __bf16* Qb  = (__bf16*)d_ws;
  __bf16* Kb  = Qb + headElems;
  __bf16* Vb  = Kb + headElems;         // stored transposed [b,h,d,t]
  __bf16* Ctx = Vb + headElems;         // [b,s,h] bf16
  // total workspace: 4 * 16 MB = 64 MB

  // 1) QKV projection (N = 6144)
  gemm_bf16_wmma<false, 1><<<dim3(24, 32), 256, 0, stream>>>(
      (const void*)hidden, w_qkv, b_qkv, nullptr, Qb, Kb, Vb);

  // 2) causal flash attention
  flash_attn_wmma<<<dim3(16, 32), 256, 0, stream>>>(Qb, Kb, Vb, Ctx);

  // 3) output projection (N = 2048), A tile via TDM
  gemm_bf16_wmma<true, 0><<<dim3(8, 32), 256, 0, stream>>>(
      (const void*)Ctx, w_dense, b_dense, out, nullptr, nullptr, nullptr);
}
